// ZSchNet_15676630630710
// MI455X (gfx1250) — matrix operand order repaired and checked by reading, not compiled, not run on hardware
//
#include <hip/hip_runtime.h>
#include <hip/hip_bf16.h>

typedef __attribute__((ext_vector_type(16))) _Float16 v16h;
typedef __attribute__((ext_vector_type(8)))  _Float16 v8h;
typedef __attribute__((ext_vector_type(8)))  float    v8f;

#define NN (4096 * 9)     // nodes
#define NE (4096 * 72)    // edges
#define NB 4096           // molecules

__device__ __forceinline__ float sspl(float x) {
  // shifted softplus, branch-free & stable:
  // log(1+e^x) - log2 = max(x,0) + log(1 + e^-|x|) - log2
  float e = __expf(-fabsf(x));          // v_exp_f32
  return fmaxf(x, 0.f) + __logf(1.f + e) - 0.6931471805599453f;
}

// ---------------------------------------------------------------------------
// Wave-level 16x64 @ 64x64 f16 WMMA GEMM from LDS.
// A: [16][64] row-major halves. Wf: fragment-swizzled weights,
//   Wf[((nt*2+kc)*32 + lane)*16 + j] = W[kc*32 + ((lane>>4)<<3) + (j&7) + ((j>>3)<<4)][nt*16 + (lane&15)]
// Out (16x64 f16) = act(A @ W + bias)
// ---------------------------------------------------------------------------
__device__ __forceinline__ void wave_gemm_16x64(
    const _Float16* __restrict__ A, const _Float16* __restrict__ Wf,
    const float* __restrict__ bias, _Float16* __restrict__ Out,
    int lane, bool act)
{
  const int m  = lane & 15;           // A row / B col / C col
  const int kh = (lane >> 4) << 3;    // A K-offset (0 or 8), also C row offset
  const v16h* W16 = (const v16h*)Wf;
#pragma unroll
  for (int nt = 0; nt < 4; ++nt) {
    v8f acc = {0.f, 0.f, 0.f, 0.f, 0.f, 0.f, 0.f, 0.f};
#pragma unroll
    for (int kc = 0; kc < 2; ++kc) {
      const v8h* ap = (const v8h*)(A + m * 64 + kc * 32);
      v8h alo = ap[kh >> 3];          // halves [kh .. kh+7]
      v8h ahi = ap[2 + (kh >> 3)];    // halves [16+kh .. 16+kh+7]
      v16h a = __builtin_shufflevector(alo, ahi,
          0, 1, 2, 3, 4, 5, 6, 7, 8, 9, 10, 11, 12, 13, 14, 15);
      v16h b = W16[(nt * 2 + kc) * 32 + lane];
      acc = __builtin_amdgcn_wmma_f32_16x16x32_f16(
          false, a, false, b, (short)0, acc, false, false);
    }
    int   col = nt * 16 + m;
    float bc  = bias[col];
#pragma unroll
    for (int r = 0; r < 8; ++r) {
      float val = acc[r] + bc;
      if (act) val = sspl(val);
      Out[(r + kh) * 64 + col] = (_Float16)val;
    }
  }
}

// ---------------------------------------------------------------------------
// Fused per-edge kernel: gaussian expansion -> g1/g2 MLP, z-embedding -> z1/z2
// MLP, msg = m[src] * (g * zl), atomic scatter-add into v[dst].
// 128 threads = 4 waves, 16 edges per wave.
// ---------------------------------------------------------------------------
__global__ __launch_bounds__(128) void k_edges(
    const float* __restrict__ pos0, const int* __restrict__ ei_src,
    const int* __restrict__ ei_dst, const int* __restrict__ zatom,
    const _Float16* __restrict__ m16, float* __restrict__ v,
    const float* g1W, const float* g1b, const float* g2W, const float* g2b,
    const float* z1W, const float* z1b, const float* z2W, const float* z2b,
    const float* c_emb)
{
  __shared__ alignas(32) _Float16 Wg1[64 * 64], Wg2[64 * 64],
                                  Wz1[64 * 64], Wz2[64 * 64];
  __shared__ float    Bg1[64], Bg2[64], Bz1[64], Bz2[64];
  __shared__ alignas(32) _Float16 bufA[4][16 * 64], bufB[4][16 * 64],
                                  bufC[4][16 * 64];
  __shared__ float    deS[4][16];
  __shared__ int      srcS[4][16], dstS[4][16];

  const int tid = threadIdx.x;
  // load weights into fragment-swizzled layout (g1 K-padded 50->64 with zeros)
  for (int i = tid; i < 64 * 64; i += 128) {
    int j  = i & 15;
    int ln = (i >> 4) & 31;
    int kc = (i >> 9) & 1;
    int nt = (i >> 10) & 3;
    int k  = kc * 32 + ((ln >> 4) << 3) + (j & 7) + ((j >> 3) << 4);
    int n  = nt * 16 + (ln & 15);
    int kn = k * 64 + n;
    Wg1[i] = (_Float16)((k < 50) ? g1W[kn] : 0.f);
    Wg2[i] = (_Float16)g2W[kn];
    Wz1[i] = (_Float16)z1W[kn];
    Wz2[i] = (_Float16)z2W[kn];
  }
  if (tid < 64) {
    Bg1[tid] = g1b[tid]; Bg2[tid] = g2b[tid];
    Bz1[tid] = z1b[tid]; Bz2[tid] = z2b[tid];
  }
  __syncthreads();

  const int wave = tid >> 5, lane = tid & 31;
  const int e0 = (blockIdx.x * 4 + wave) * 16;

  // phase 1: edge distances + endpoints
  if (lane < 16) {
    int e = e0 + lane;
    int s = ei_src[e], d = ei_dst[e];
    float dx = pos0[d * 3 + 0] - pos0[s * 3 + 0];
    float dy = pos0[d * 3 + 1] - pos0[s * 3 + 1];
    float dz = pos0[d * 3 + 2] - pos0[s * 3 + 2];
    deS[wave][lane]  = sqrtf(dx * dx + dy * dy + dz * dz);
    srcS[wave][lane] = s;
    dstS[wave][lane] = d;
  }
  __syncthreads();

  // phase 2: gaussian basis (cols 50..63 zero-padded)
  for (int i = lane; i < 16 * 64; i += 32) {
    int r = i >> 6, c = i & 63;
    float val = 0.f;
    if (c < 50) {
      float td = deS[wave][r] - 0.1f * (float)c;
      val = __expf(-10.f * td * td);
    }
    bufA[wave][i] = (_Float16)val;
  }
  __syncthreads();

  // g-branch: bufA -> bufB -> bufA (sspl both)
  wave_gemm_16x64(bufA[wave], Wg1, Bg1, bufB[wave], lane, true);
  __syncthreads();
  wave_gemm_16x64(bufB[wave], Wg2, Bg2, bufA[wave], lane, true);
  __syncthreads();

  // z-branch: zz = c_emb[z_i] * c_emb[z_j]
  for (int i = lane; i < 16 * 64; i += 32) {
    int r = i >> 6, c = i & 63;
    int zi = zatom[dstS[wave][r]], zj = zatom[srcS[wave][r]];
    bufB[wave][i] = (_Float16)(c_emb[zi * 64 + c] * c_emb[zj * 64 + c]);
  }
  __syncthreads();
  wave_gemm_16x64(bufB[wave], Wz1, Bz1, bufC[wave], lane, true);
  __syncthreads();
  wave_gemm_16x64(bufC[wave], Wz2, Bz2, bufB[wave], lane, true);
  __syncthreads();

  // msg = m16[src] * (g * zl); segment-sum via f32 atomics
  for (int i = lane; i < 16 * 64; i += 32) {
    int r = i >> 6, c = i & 63;
    float w   = (float)bufA[wave][i] * (float)bufB[wave][i];
    float msg = (float)m16[srcS[wave][r] * 64 + c] * w;
    atomicAdd(&v[dstS[wave][r] * 64 + c], msg);
  }
}

// ---------------------------------------------------------------------------
// Generic WMMA GEMM: Out = act(In @ W + b). In is [M, KD] f32 (pre-padded),
// W is [KSRC, NO] f32 (rows >= KSRC treated as zero), stored in LDS in
// fragment-swizzled layout. Each wave: one 16-row tile.
// ACT = shifted softplus. ACC = accumulate into OutF. OutH optional f16 copy.
// ---------------------------------------------------------------------------
template <int KD, int KSRC, int NO, int WAVES, bool ACT, bool ACC>
__global__ __launch_bounds__(WAVES * 32) void k_gemm(
    const float* __restrict__ In, const float* __restrict__ Wg,
    const float* __restrict__ bg, float* OutF, _Float16* OutH, int M)
{
  constexpr int KC = KD / 32;   // k chunks
  __shared__ alignas(32) _Float16 Wl[KD * NO];
  __shared__ float    bl[NO];
  __shared__ alignas(32) _Float16 Al[WAVES][16 * KD];

  const int tid = threadIdx.x;
  for (int i = tid; i < KD * NO; i += WAVES * 32) {
    int j  = i & 15;
    int ln = (i >> 4) & 31;
    int kc = (i >> 9) % KC;
    int nt = (i >> 9) / KC;
    int k  = kc * 32 + ((ln >> 4) << 3) + (j & 7) + ((j >> 3) << 4);
    int n  = nt * 16 + (ln & 15);
    Wl[i] = (_Float16)((k < KSRC) ? Wg[k * NO + n] : 0.f);
  }
  for (int i = tid; i < NO; i += WAVES * 32) bl[i] = bg[i];

  const int wave = tid >> 5, lane = tid & 31;
  const int rbase = (blockIdx.x * WAVES + wave) * 16;
  for (int i = lane; i < 16 * KD; i += 32) {
    int r = i / KD, c = i % KD;
    int row = rbase + r;
    Al[wave][i] = (_Float16)((row < M) ? In[(size_t)row * KD + c] : 0.f);
  }
  __syncthreads();

  const _Float16* A = Al[wave];
  const v16h* W16 = (const v16h*)Wl;
  const int m  = lane & 15;
  const int kh = (lane >> 4) << 3;
#pragma unroll
  for (int nt = 0; nt < NO / 16; ++nt) {
    v8f acc = {0.f, 0.f, 0.f, 0.f, 0.f, 0.f, 0.f, 0.f};
#pragma unroll
    for (int kc = 0; kc < KC; ++kc) {
      const v8h* ap = (const v8h*)(A + m * KD + kc * 32);
      v8h alo = ap[kh >> 3];
      v8h ahi = ap[2 + (kh >> 3)];
      v16h a = __builtin_shufflevector(alo, ahi,
          0, 1, 2, 3, 4, 5, 6, 7, 8, 9, 10, 11, 12, 13, 14, 15);
      v16h b = W16[(nt * KC + kc) * 32 + lane];
      acc = __builtin_amdgcn_wmma_f32_16x16x32_f16(
          false, a, false, b, (short)0, acc, false, false);
    }
    int   col = nt * 16 + m;
    float bc  = bl[col];
#pragma unroll
    for (int r = 0; r < 8; ++r) {
      int row = rbase + r + kh;
      if (row < M) {
        float val = acc[r] + bc;
        if (ACT) val = sspl(val);
        size_t off = (size_t)row * NO + col;
        if (OutF) { if (ACC) OutF[off] += val; else OutF[off] = val; }
        if (OutH) OutH[off] = (_Float16)val;
      }
    }
  }
}

// ---------------------------------------------------------------------------
// Small elementwise / gather kernels
// ---------------------------------------------------------------------------
__global__ void k_embed(const float* __restrict__ emb, const int* __restrict__ z,
                        float* __restrict__ x) {
  int i = blockIdx.x * 256 + threadIdx.x;
  if (i < NN * 64) x[i] = emb[z[i >> 6] * 64 + (i & 63)];
}

__global__ void k_gather(const float* __restrict__ x, const int* __restrict__ nuc,
                         float* __restrict__ xg) {
  int i = blockIdx.x * 256 + threadIdx.x;
  if (i < NB * 64) xg[i] = x[(size_t)(nuc[i >> 6] - 1) * 64 + (i & 63)];
}

__global__ void k_solvin(const float* __restrict__ pos0, const float* __restrict__ angle,
                         const float* __restrict__ morse, float* __restrict__ sv) {
  int idx = blockIdx.x * 256 + threadIdx.x;
  if (idx >= NB * 64) return;
  int b = idx >> 6, s = idx & 63;
  float val = 0.f;
  if (s < 8) {
    val = angle[b * 8 + s];
  } else if (s < 44) {
    int p = s - 8;                     // upper-triangular pair index of 9x9
    int i = 0, acc = 0;
    while (acc + (8 - i) <= p) { acc += 8 - i; ++i; }
    int j = i + 1 + (p - acc);
    const float* pi = pos0 + (size_t)(b * 9 + i) * 3;
    const float* pj = pos0 + (size_t)(b * 9 + j) * 3;
    float dx = pi[0] - pj[0], dy = pi[1] - pj[1], dz = pi[2] - pj[2];
    val = sqrtf(dx * dx + dy * dy + dz * dz);
  } else if (s < 52) {
    val = morse[b * 8 + (s - 44)];
  }
  sv[idx] = val;                       // cols 52..63 zero-padded
}

__global__ void k_concat(const float* __restrict__ xn, const float* __restrict__ sv,
                         float* __restrict__ fin) {
  int i = blockIdx.x * 256 + threadIdx.x;
  if (i >= NB * 192) return;
  int b = i / 192, c = i % 192;
  fin[i] = (c < 64) ? xn[b * 64 + c] : sv[b * 128 + (c - 64)];
}

__global__ void k_final(const float* __restrict__ h2, const float* __restrict__ W,
                        const float* __restrict__ b, float* __restrict__ out, int M) {
  int i = blockIdx.x * 256 + threadIdx.x;
  if (i >= M) return;
  float s = b[0];
#pragma unroll
  for (int k = 0; k < 32; ++k) s += h2[i * 32 + k] * W[k];
  out[i] = s;
}

// ---------------------------------------------------------------------------
extern "C" void kernel_launch(void* const* d_in, const int* in_sizes, int n_in,
                              void* d_out, int out_size, void* d_ws, size_t ws_size,
                              hipStream_t stream) {
  (void)in_sizes; (void)out_size; (void)ws_size;
  if (n_in < 55) return;

  const float* pos0  = (const float*)d_in[0];
  const int*   z     = (const int*)d_in[1];
  const int*   ei    = (const int*)d_in[2];   // [2, E]: src row then dst row
  const float* angle = (const float*)d_in[3];
  const float* morse = (const float*)d_in[6]; // keylen (4) / hook (5) unused
  const int*   nuc   = (const int*)d_in[7];
  auto F = [&](int i) { return (const float*)d_in[i]; };
  const float* emb_z = F(8);

  char* ws = (char*)d_ws;
  size_t off = 0;
  auto alloc = [&](size_t bytes) {
    size_t o = off; off += (bytes + 255) & ~(size_t)255; return o;
  };
  float*    x    = (float*)(ws + alloc((size_t)NN * 64 * 4));
  float*    v    = (float*)(ws + alloc((size_t)NN * 64 * 4));
  _Float16* m16  = (_Float16*)(ws + alloc((size_t)NN * 64 * 2));
  float*    xg   = (float*)(ws + alloc((size_t)NB * 64 * 4));
  float*    sv   = (float*)(ws + alloc((size_t)NB * 64 * 4));
  float*    solv = (float*)(ws + alloc((size_t)NB * 128 * 4));
  float*    fin  = (float*)(ws + alloc((size_t)NB * 192 * 4));
  float*    h1   = (float*)(ws + alloc((size_t)NB * 128 * 4));
  float*    h2   = (float*)(ws + alloc((size_t)NB * 32 * 4));
  float*    out  = (float*)d_out;

  // node embedding
  k_embed<<<(NN * 64 + 255) / 256, 256, 0, stream>>>(emb_z, z, x);

  // interaction blocks
  for (int t = 0; t < 2; ++t) {
    int ib = 9 + t * 15;  // c_emb,g1{W,b},g2{W,b},lin1{W,b},mlp1{W,b},mlp2{W,b},z1{W,b},z2{W,b}
    const float* c_emb = F(ib + 0);
    const float *g1W = F(ib + 1), *g1b = F(ib + 2), *g2W = F(ib + 3), *g2b = F(ib + 4);
    const float *l1W = F(ib + 5), *l1b = F(ib + 6);
    const float *m1W = F(ib + 7), *m1b = F(ib + 8), *m2W = F(ib + 9), *m2b = F(ib + 10);
    const float *z1W = F(ib + 11), *z1b = F(ib + 12), *z2W = F(ib + 13), *z2b = F(ib + 14);

    hipMemsetAsync(v, 0, (size_t)NN * 64 * 4, stream);
    // m = lin1(x) -> f16 for edge gathers
    k_gemm<64, 64, 64, 4, false, false><<<NN / 64, 128, 0, stream>>>(
        x, l1W, l1b, nullptr, m16, NN);
    // fused edge pipeline + scatter-add
    k_edges<<<NE / 64, 128, 0, stream>>>(pos0, ei, ei + NE, z, m16, v,
        g1W, g1b, g2W, g2b, z1W, z1b, z2W, z2b, c_emb);
    // v = sspl(mlp1(v)) in place; x += mlp2(v)
    k_gemm<64, 64, 64, 4, true, false><<<NN / 64, 128, 0, stream>>>(
        v, m1W, m1b, v, nullptr, NN);
    k_gemm<64, 64, 64, 4, false, true><<<NN / 64, 128, 0, stream>>>(
        v, m2W, m2b, x, nullptr, NN);
  }

  // post MLP only on the gathered nucleophile rows (saves 9x work)
  k_gather<<<(NB * 64 + 255) / 256, 256, 0, stream>>>(x, nuc, xg);
  k_gemm<64, 64, 64, 4, true, false><<<NB / 64, 128, 0, stream>>>(
      xg, F(39), F(40), xg, nullptr, NB);
  k_gemm<64, 64, 64, 4, false, false><<<NB / 64, 128, 0, stream>>>(
      xg, F(41), F(42), xg, nullptr, NB);

  // solvent branch: [angle|dist36|morse] (52, padded to 64) -> 64 -> sspl -> 128
  k_solvin<<<(NB * 64 + 255) / 256, 256, 0, stream>>>(pos0, angle, morse, sv);
  k_gemm<64, 52, 64, 4, false, false><<<NB / 64, 128, 0, stream>>>(
      sv, F(49), F(50), sv, nullptr, NB);
  k_gemm<64, 64, 64, 4, true, false><<<NB / 64, 128, 0, stream>>>(
      sv, F(51), F(52), sv, nullptr, NB);
  k_gemm<64, 64, 128, 4, false, false><<<NB / 64, 128, 0, stream>>>(
      sv, F(53), F(54), solv, nullptr, NB);

  // head: concat(192) -> 128 -> 32 -> 1
  k_concat<<<(NB * 192 + 255) / 256, 256, 0, stream>>>(xg, solv, fin);
  k_gemm<192, 192, 128, 2, true, false><<<NB / 32, 64, 0, stream>>>(
      fin, F(43), F(44), h1, nullptr, NB);
  k_gemm<128, 128, 32, 4, true, false><<<NB / 64, 128, 0, stream>>>(
      h1, F(45), F(46), h2, nullptr, NB);
  k_final<<<(NB + 255) / 256, 256, 0, stream>>>(h2, F(47), F(48), out, NB);
}
